// GraphBatchNetAMP_16604343566363
// MI455X (gfx1250) — matrix operand order repaired and checked by reading, not compiled, not run on hardware
//
#include <hip/hip_runtime.h>
#include <hip/hip_bf16.h>

typedef __attribute__((ext_vector_type(16))) _Float16 v16h;
typedef __attribute__((ext_vector_type(8)))  _Float16 v8h;
typedef __attribute__((ext_vector_type(8)))  float    v8f;

#define BB 4
#define NN 10000
#define MM 320000
#define FXc 64
#define FEc 16
#define HHc 64

#define NBLK_N 32    // blocks per graph, node kernel
#define NBLK_E 125   // blocks per graph, edge kernel

static __device__ __forceinline__ v8f wmma16(v16h a, v16h b, v8f c) {
  return __builtin_amdgcn_wmma_f32_16x16x32_f16(false, a, false, b, (short)0, c, false, false);
}

static __device__ __forceinline__ v8f zero8() {
  v8f z;
  #pragma unroll
  for (int i = 0; i < 8; ++i) z[i] = 0.f;
  return z;
}

// A fragment (16x32 f16) for row `rowp` of a f32 matrix; chunk base in floats.
// a[j] = A[m][(j/8)*16 + half*8 + (j%8)]  (m = lane%16 handled by caller's rowp)
static __device__ __forceinline__ v16h afrag_g(const float* __restrict__ rowp, int base, int half) {
  const float* p = rowp + base + half * 8;
  v16h r;
  #pragma unroll
  for (int j = 0; j < 8; ++j) {
    r[j]     = (_Float16)p[j];
    r[j + 8] = (_Float16)p[16 + j];
  }
  return r;
}

// A fragment for the E chunk (16 valid K, padded to 32). Also returns the raw
// 8 f32 values (reused for the gate*e accumulation).
static __device__ __forceinline__ v16h afrag_e(const float* __restrict__ erow, int half, float* ev) {
  v16h r;
  #pragma unroll
  for (int j = 0; j < 8; ++j) {
    float x = erow[half * 8 + j];
    ev[j] = x;
    r[j] = (_Float16)x;
    r[j + 8] = (_Float16)0.f;
  }
  return r;
}

// B fragment (32x16 f16): b[j] = W[kbase + j + (j/8)*8][col], kbase = kc*32 + half*8
static __device__ __forceinline__ v16h bfrag_g(const float* __restrict__ W, int ld, int kbase, int col) {
  v16h r;
  #pragma unroll
  for (int j = 0; j < 16; ++j) r[j] = (_Float16)W[(size_t)(kbase + j + (j / 8) * 8) * ld + col];
  return r;
}

// B fragment from LDS-resident transposed f16 weights: Wt[col*stride + k]
static __device__ __forceinline__ v16h bfrag_lds(const _Float16* Wt, int stride, int col, int kbase) {
  const v8h* p = (const v8h*)(Wt + col * stride + kbase);
  v8h x0 = p[0], x1 = p[2];
  return __builtin_shufflevector(x0, x1, 0,1,2,3,4,5,6,7,8,9,10,11,12,13,14,15);
}

// A fragment from per-wave LDS h1 tile (row-major f16, stride 72 halves)
static __device__ __forceinline__ v16h afrag_lds(const _Float16* hrow, int base, int half) {
  const v8h* p1 = (const v8h*)(hrow + base + half * 8);
  const v8h* p2 = (const v8h*)(hrow + base + 16 + half * 8);
  v8h x0 = p1[0], x1 = p2[0];
  return __builtin_shufflevector(x0, x1, 0,1,2,3,4,5,6,7,8,9,10,11,12,13,14,15);
}

// ---------------- init: zero accumulators -------------------
__global__ void k_init(float* ws) {
  int i = blockIdx.x * blockDim.x + threadIdx.x;
  if (i < BB * (64 + 64 + 16 + 1)) ws[i] = 0.f;
}

// ---------------- node MLP: accumulate sum_n hx -------------
__global__ void __launch_bounds__(256) k_node(const float* __restrict__ X,
                                              const float* __restrict__ Wn1, const float* __restrict__ bn1,
                                              const float* __restrict__ Wn2, const float* __restrict__ bn2,
                                              float* __restrict__ node_acc) {
  __shared__ _Float16 h1buf[8][16 * 72];
  const int lane = threadIdx.x & 31;
  const int wv   = threadIdx.x >> 5;
  const int half = lane >> 4;
  const int l15  = lane & 15;
  const int b    = blockIdx.x / NBLK_N;
  const int wg   = (blockIdx.x % NBLK_N) * 8 + wv;
  const int WPG  = NBLK_N * 8;

  v16h bf1[2][4], bf2[2][4];
  #pragma unroll
  for (int kc = 0; kc < 2; ++kc)
    #pragma unroll
    for (int nt = 0; nt < 4; ++nt) {
      bf1[kc][nt] = bfrag_g(Wn1, HHc, kc * 32 + half * 8, nt * 16 + l15);
      bf2[kc][nt] = bfrag_g(Wn2, HHc, kc * 32 + half * 8, nt * 16 + l15);
    }
  float bias1[4], bias2[4];
  #pragma unroll
  for (int nt = 0; nt < 4; ++nt) { bias1[nt] = bn1[nt * 16 + l15]; bias2[nt] = bn2[nt * 16 + l15]; }

  v8f hsum[4];
  #pragma unroll
  for (int nt = 0; nt < 4; ++nt) hsum[nt] = zero8();

  const float* Xb = X + (size_t)b * NN * FXc;
  _Float16* hb = h1buf[wv];

  for (int t = wg; t < NN / 16; t += WPG) {
    const float* rowp = Xb + (size_t)(t * 16 + l15) * FXc;
    v8f h1a[4];
    #pragma unroll
    for (int nt = 0; nt < 4; ++nt) h1a[nt] = zero8();
    #pragma unroll
    for (int kc = 0; kc < 2; ++kc) {
      v16h a = afrag_g(rowp, kc * 32, half);
      #pragma unroll
      for (int nt = 0; nt < 4; ++nt) h1a[nt] = wmma16(a, bf1[kc][nt], h1a[nt]);
    }
    #pragma unroll
    for (int nt = 0; nt < 4; ++nt)
      #pragma unroll
      for (int v = 0; v < 8; ++v)
        hb[(v + 8 * half) * 72 + nt * 16 + l15] = (_Float16)fmaxf(h1a[nt][v] + bias1[nt], 0.f);

    v8f h2a[4];
    #pragma unroll
    for (int nt = 0; nt < 4; ++nt) h2a[nt] = zero8();
    const _Float16* hrow = hb + l15 * 72;
    #pragma unroll
    for (int kc = 0; kc < 2; ++kc) {
      v16h a = afrag_lds(hrow, kc * 32, half);
      #pragma unroll
      for (int nt = 0; nt < 4; ++nt) h2a[nt] = wmma16(a, bf2[kc][nt], h2a[nt]);
    }
    #pragma unroll
    for (int nt = 0; nt < 4; ++nt)
      #pragma unroll
      for (int v = 0; v < 8; ++v)
        hsum[nt][v] += fmaxf(h2a[nt][v] + bias2[nt], 0.f);
  }

  #pragma unroll
  for (int nt = 0; nt < 4; ++nt) {
    float s = 0.f;
    #pragma unroll
    for (int v = 0; v < 8; ++v) s += hsum[nt][v];
    s += __shfl_xor(s, 16, 32);
    if (lane < 16) atomicAdd(&node_acc[b * 64 + nt * 16 + l15], s);
  }
}

// -------- edge MLP: accumulate sum_e msg, sum_e gate*e, sum_e gate --------
__global__ void __launch_bounds__(256) k_edge(const float* __restrict__ X, const float* __restrict__ E,
                                              const int* __restrict__ EI,
                                              const float* __restrict__ We1, const float* __restrict__ be1,
                                              const float* __restrict__ We2, const float* __restrict__ be2,
                                              const float* __restrict__ gsp, const float* __restrict__ gcp,
                                              float* __restrict__ msg_acc, float* __restrict__ se_acc,
                                              float* __restrict__ sg_acc) {
  __shared__ _Float16 W1t[64 * 168];        // We1 transposed, K padded 144->160, stride 168
  __shared__ _Float16 h1buf[8][16 * 72];
  const int tid = threadIdx.x;

  for (int i = tid; i < 64 * 160; i += 256) {
    int col = i / 160, k = i % 160;
    float v = (k < 144) ? We1[(size_t)k * 64 + col] : 0.f;
    W1t[col * 168 + k] = (_Float16)v;
  }
  __syncthreads();

  const int lane = tid & 31;
  const int wv   = tid >> 5;
  const int half = lane >> 4;
  const int l15  = lane & 15;
  const int b    = blockIdx.x / NBLK_E;
  const int wg   = (blockIdx.x % NBLK_E) * 8 + wv;
  const int WPG  = NBLK_E * 8;
  const float gs = gsp[0], gc = gcp[0];

  v16h b2f[2][4];
  #pragma unroll
  for (int kc = 0; kc < 2; ++kc)
    #pragma unroll
    for (int nt = 0; nt < 4; ++nt)
      b2f[kc][nt] = bfrag_g(We2, HHc, kc * 32 + half * 8, nt * 16 + l15);
  float bias1[4], bias2[4];
  #pragma unroll
  for (int nt = 0; nt < 4; ++nt) { bias1[nt] = be1[nt * 16 + l15]; bias2[nt] = be2[nt * 16 + l15]; }

  v8f msum[4];
  #pragma unroll
  for (int nt = 0; nt < 4; ++nt) msum[nt] = zero8();
  float acc_e[8];
  #pragma unroll
  for (int j = 0; j < 8; ++j) acc_e[j] = 0.f;
  float acc_g = 0.f;

  const float* Xb = X + (size_t)b * NN * FXc;
  const float* Eb = E + (size_t)b * MM * FEc;
  const int*   Ib = EI + (size_t)b * MM * 2;
  _Float16* hb = h1buf[wv];

  for (int t = wg; t < MM / 16; t += WPG) {
    const int edge = t * 16 + l15;
    const int src = Ib[(size_t)edge * 2 + 0];
    const int dst = Ib[(size_t)edge * 2 + 1];
    const float* erow = Eb + (size_t)edge * FEc;

    float z  = gs * (erow[2] - gc);
    float sp = (z > 20.f) ? z : log1pf(__expf(z));
    float gate = sp + 0.001f;
    float g[8];
    #pragma unroll
    for (int v = 0; v < 8; ++v) g[v] = __shfl(gate, v + 8 * half, 32);

    const float* xs = Xb + (size_t)src * FXc;
    const float* xd = Xb + (size_t)dst * FXc;

    v8f h1a[4];
    #pragma unroll
    for (int nt = 0; nt < 4; ++nt) h1a[nt] = zero8();

    #pragma unroll
    for (int kc = 0; kc < 4; ++kc) {       // K chunks 0..3: x_src, x_dst
      const float* rp = (kc < 2) ? xs : xd;
      v16h a = afrag_g(rp, (kc & 1) * 32, half);
      #pragma unroll
      for (int nt = 0; nt < 4; ++nt)
        h1a[nt] = wmma16(a, bfrag_lds(W1t, 168, nt * 16 + l15, kc * 32 + half * 8), h1a[nt]);
    }
    float ev[8];
    {                                       // K chunk 4: e (padded 16->32)
      v16h a = afrag_e(erow, half, ev);
      #pragma unroll
      for (int nt = 0; nt < 4; ++nt)
        h1a[nt] = wmma16(a, bfrag_lds(W1t, 168, nt * 16 + l15, 128 + half * 8), h1a[nt]);
    }

    #pragma unroll
    for (int nt = 0; nt < 4; ++nt)
      #pragma unroll
      for (int v = 0; v < 8; ++v)
        hb[(v + 8 * half) * 72 + nt * 16 + l15] = (_Float16)fmaxf(h1a[nt][v] + bias1[nt], 0.f);

    v8f h2a[4];
    #pragma unroll
    for (int nt = 0; nt < 4; ++nt) h2a[nt] = zero8();
    const _Float16* hrow = hb + l15 * 72;
    #pragma unroll
    for (int kc = 0; kc < 2; ++kc) {
      v16h a = afrag_lds(hrow, kc * 32, half);
      #pragma unroll
      for (int nt = 0; nt < 4; ++nt) h2a[nt] = wmma16(a, b2f[kc][nt], h2a[nt]);
    }

    #pragma unroll
    for (int nt = 0; nt < 4; ++nt)
      #pragma unroll
      for (int v = 0; v < 8; ++v)
        msum[nt][v] += g[v] * fmaxf(h2a[nt][v] + bias2[nt], 0.f);

    #pragma unroll
    for (int j = 0; j < 8; ++j) acc_e[j] += gate * ev[j];
    if (lane < 16) acc_g += gate;
  }

  // wave reductions + a few f32 atomics
  #pragma unroll
  for (int nt = 0; nt < 4; ++nt) {
    float s = 0.f;
    #pragma unroll
    for (int v = 0; v < 8; ++v) s += msum[nt][v];
    s += __shfl_xor(s, 16, 32);
    if (lane < 16) atomicAdd(&msg_acc[b * 64 + nt * 16 + l15], s);
  }
  #pragma unroll
  for (int j = 0; j < 8; ++j) {
    float s = acc_e[j];
    s += __shfl_xor(s, 8, 32);
    s += __shfl_xor(s, 4, 32);
    s += __shfl_xor(s, 2, 32);
    s += __shfl_xor(s, 1, 32);
    if (l15 == 0) atomicAdd(&se_acc[b * 16 + half * 8 + j], s);
  }
  {
    float s = acc_g;
    s += __shfl_xor(s, 8, 32);
    s += __shfl_xor(s, 4, 32);
    s += __shfl_xor(s, 2, 32);
    s += __shfl_xor(s, 1, 32);
    if (lane == 0) atomicAdd(&sg_acc[b], s);
  }
}

// ---------------- final readout (tiny) ----------------------
__global__ void k_final(const float* __restrict__ ws,
                        const float* __restrict__ Wp, const float* __restrict__ bp,
                        const float* __restrict__ Wr1, const float* __restrict__ br1,
                        const float* __restrict__ Wr2, const float* __restrict__ br2,
                        float* __restrict__ out) {
  __shared__ float ctx[128];
  __shared__ float hh[64];
  const int b = blockIdx.x, t = threadIdx.x;   // 64 threads
  const float* node_acc = ws;
  const float* msg_acc  = ws + BB * 64;
  const float* se       = ws + BB * 128;
  const float* sg       = ws + BB * 128 + BB * 16;

  float nc = (node_acc[b * 64 + t] + 2.f * msg_acc[b * 64 + t]) / (float)NN;
  float ec = sg[b] * bp[t];
  #pragma unroll
  for (int k = 0; k < 16; ++k) ec += se[b * 16 + k] * Wp[k * 64 + t];
  ec /= ((float)MM + 1e-6f);
  ctx[t] = nc;
  ctx[64 + t] = ec;
  __syncthreads();

  float h = br1[t];
  for (int j = 0; j < 128; ++j) h += ctx[j] * Wr1[j * 64 + t];
  hh[t] = fmaxf(h, 0.f);
  __syncthreads();

  if (t < 2) {
    float o = br2[t];
    for (int j = 0; j < 64; ++j) o += hh[j] * Wr2[j * 2 + t];
    out[b * 2 + t] = o;
  }
}

extern "C" void kernel_launch(void* const* d_in, const int* in_sizes, int n_in,
                              void* d_out, int out_size, void* d_ws, size_t ws_size,
                              hipStream_t stream) {
  (void)in_sizes; (void)n_in; (void)out_size; (void)ws_size;
  const float* X   = (const float*)d_in[0];
  const float* E   = (const float*)d_in[1];
  const int*   EI  = (const int*)  d_in[2];
  const float* Wn1 = (const float*)d_in[3];
  const float* bn1 = (const float*)d_in[4];
  const float* Wn2 = (const float*)d_in[5];
  const float* bn2 = (const float*)d_in[6];
  const float* We1 = (const float*)d_in[7];
  const float* be1 = (const float*)d_in[8];
  const float* We2 = (const float*)d_in[9];
  const float* be2 = (const float*)d_in[10];
  const float* Wp  = (const float*)d_in[11];
  const float* bp  = (const float*)d_in[12];
  const float* Wr1 = (const float*)d_in[13];
  const float* br1 = (const float*)d_in[14];
  const float* Wr2 = (const float*)d_in[15];
  const float* br2 = (const float*)d_in[16];
  const float* gsp = (const float*)d_in[17];
  const float* gcp = (const float*)d_in[18];

  float* ws  = (float*)d_ws;
  float* out = (float*)d_out;

  k_init <<<3, 256, 0, stream>>>(ws);
  k_node <<<BB * NBLK_N, 256, 0, stream>>>(X, Wn1, bn1, Wn2, bn2, ws);
  k_edge <<<BB * NBLK_E, 256, 0, stream>>>(X, E, EI, We1, be1, We2, be2, gsp, gcp,
                                           ws + BB * 64, ws + BB * 128, ws + BB * 128 + BB * 16);
  k_final<<<BB, 64, 0, stream>>>(ws, Wp, bp, Wr1, br1, Wr2, br2, out);
}
